// GQA_9500467658942
// MI455X (gfx1250) — matrix operand order repaired and checked by reading, not compiled, not run on hardware
//
#include <hip/hip_runtime.h>

typedef __bf16 bf16_t;
typedef __bf16 v16bf __attribute__((ext_vector_type(16)));
typedef float  v8f   __attribute__((ext_vector_type(8)));

constexpr int B_ = 2, T_ = 2048, C_ = 2048, NH_ = 32, NKVH_ = 8, HS_ = 64;
constexpr int NQKV_ = C_ + 2 * NKVH_ * HS_;   // 3072
constexpr int REP_ = NH_ / NKVH_;             // 4

#if defined(__HIP_DEVICE_COMPILE__) && \
    __has_builtin(__builtin_amdgcn_global_load_async_to_lds_b128) && \
    __has_builtin(__builtin_amdgcn_s_wait_asynccnt)
#define USE_ASYNC_LDS 1
#else
#define USE_ASYNC_LDS 0
#endif

#if USE_ASYNC_LDS
typedef int v4i_ __attribute__((ext_vector_type(4)));
typedef __attribute__((address_space(1))) v4i_ g_v4i;
typedef __attribute__((address_space(3))) v4i_ l_v4i;
__device__ __forceinline__ void async_cp_b128(const unsigned short* g, unsigned short* l) {
    __builtin_amdgcn_global_load_async_to_lds_b128((g_v4i*)g, (l_v4i*)l, 0, 0);
}
#endif

__device__ __forceinline__ unsigned short f2bf_bits(float x) {
    __bf16 b = (__bf16)x;
    return __builtin_bit_cast(unsigned short, b);
}

// ---------------------------------------------------------------------------
// fp32 -> bf16 elementwise convert
// ---------------------------------------------------------------------------
__global__ __launch_bounds__(256)
void f32_to_bf16_kernel(const float* __restrict__ in, bf16_t* __restrict__ out, size_t n) {
    size_t i = (size_t)blockIdx.x * 256 + threadIdx.x;
    if (i < n) out[i] = (bf16_t)in[i];
}

// ---------------------------------------------------------------------------
// Tiled bf16 WMMA GEMM:  C[M,N] = A[M,K] * B[K,N] + bias[N]   (f32 out)
// Block tile 128(M) x 128(N), 8 waves (256 threads), K step 32,
// double-buffered LDS, async global->LDS staging for the A tile.
// Wave (wid&3, wid>>2) computes 32x64 via 2x4 accumulators of 16x16.
// ---------------------------------------------------------------------------
__global__ __launch_bounds__(256)
void gemm_bf16_wmma(const bf16_t* __restrict__ A, const bf16_t* __restrict__ Bm,
                    const float* __restrict__ bias, float* __restrict__ Cout,
                    int M, int N, int K) {
    __shared__ __align__(16) unsigned short sA[2][128 * 32];   // [m][k]
    __shared__ __align__(16) unsigned short sB[2][128 * 32];   // [n][k] (transposed)

    const int tid  = threadIdx.x;
    const int lane = tid & 31;
    const int wid  = tid >> 5;
    const int lh   = lane & 15;
    const int hi   = lane >> 4;
    const int bm   = blockIdx.x * 128;
    const int bn   = blockIdx.y * 128;
    const int wm   = (wid & 3) * 32;
    const int wn   = (wid >> 2) * 64;

    v8f acc[2][4];
    #pragma unroll
    for (int am = 0; am < 2; ++am)
        #pragma unroll
        for (int nt = 0; nt < 4; ++nt)
            #pragma unroll
            for (int j = 0; j < 8; ++j) acc[am][nt][j] = 0.0f;

    // ---- stagers -----------------------------------------------------------
    // A tile: 128 rows x 32 halves; 2 b128 segments per thread
    auto stageA = [&](int k0, unsigned short* buf) {
        #pragma unroll
        for (int s = 0; s < 2; ++s) {
            const int row = (tid >> 2) + s * 64;
            const int col = (tid & 3) * 8;
            const unsigned short* g =
                (const unsigned short*)A + (size_t)(bm + row) * K + k0 + col;
#if USE_ASYNC_LDS
            async_cp_b128(g, &buf[row * 32 + col]);
#else
            *(uint4*)&buf[row * 32 + col] = *(const uint4*)g;
#endif
        }
    };
    // B tile: 32 rows x 128 halves; 2 b128 segments per thread, via registers
    auto loadB = [&](int k0, uint4 br[2]) {
        #pragma unroll
        for (int s = 0; s < 2; ++s)
            br[s] = *(const uint4*)((const unsigned short*)Bm +
                     (size_t)(k0 + (tid >> 4) * 2 + s) * N + bn + (tid & 15) * 8);
    };
    auto storeB = [&](unsigned short* buf, const uint4 br[2]) {
        #pragma unroll
        for (int s = 0; s < 2; ++s) {
            union { uint4 q; unsigned short hh[8]; } u;
            u.q = br[s];
            const int kr = (tid >> 4) * 2 + s;
            const int nc = (tid & 15) * 8;
            #pragma unroll
            for (int e = 0; e < 8; ++e) buf[(nc + e) * 32 + kr] = u.hh[e];
        }
    };
    auto compute = [&](int c) {
        v16bf af[2];
        #pragma unroll
        for (int am = 0; am < 2; ++am) {
            union { uint4 q[2]; v16bf v; } ua;
            const unsigned short* p = &sA[c][(wm + am * 16 + lh) * 32 + hi * 8];
            ua.q[0] = *(const uint4*)p;
            ua.q[1] = *(const uint4*)(p + 16);
            af[am] = ua.v;
        }
        #pragma unroll
        for (int nt = 0; nt < 4; ++nt) {
            union { uint4 q[2]; v16bf v; } ub;
            const unsigned short* p = &sB[c][(wn + nt * 16 + lh) * 32 + hi * 8];
            ub.q[0] = *(const uint4*)p;
            ub.q[1] = *(const uint4*)(p + 16);
            acc[0][nt] = __builtin_amdgcn_wmma_f32_16x16x32_bf16(
                false, af[0], false, ub.v, (short)0, acc[0][nt], false, false);
            acc[1][nt] = __builtin_amdgcn_wmma_f32_16x16x32_bf16(
                false, af[1], false, ub.v, (short)0, acc[1][nt], false, false);
        }
    };

    // ---- pipeline ----------------------------------------------------------
    uint4 br[2];
    loadB(0, br);
    stageA(0, sA[0]);
    storeB(sB[0], br);
#if USE_ASYNC_LDS
    __builtin_amdgcn_s_wait_asynccnt(0);
#endif
    __syncthreads();

    int cur = 0;
    for (int k0 = 32; k0 < K; k0 += 32) {
        const int nxt = cur ^ 1;
        stageA(k0, sA[nxt]);       // async: overlaps with WMMA below
        loadB(k0, br);             // global loads overlap with WMMA below
        compute(cur);
        storeB(sB[nxt], br);
#if USE_ASYNC_LDS
        __builtin_amdgcn_s_wait_asynccnt(0);
#endif
        __syncthreads();
        cur = nxt;
    }
    compute(cur);

    // ---- epilogue ----------------------------------------------------------
    #pragma unroll
    for (int am = 0; am < 2; ++am)
        #pragma unroll
        for (int nt = 0; nt < 4; ++nt)
            #pragma unroll
            for (int j = 0; j < 8; ++j) {
                int row = bm + wm + am * 16 + j + hi * 8;  // C layout: M=j+8*(lane/16)
                int col = bn + wn + nt * 16 + lh;          // N = lane%16
                Cout[(size_t)row * N + col] = acc[am][nt][j] + bias[col];
            }
}

// ---------------------------------------------------------------------------
// RoPE + scatter: qkv f32 [B*T, 3072] -> q/k/v bf16 head-major
//   q: [B,NH,T,HS]   k,v: [B,NKVH,T,HS]
// ---------------------------------------------------------------------------
__global__ __launch_bounds__(256)
void rope_scatter(const float* __restrict__ qkv, const float* __restrict__ fcos,
                  const float* __restrict__ fsin, bf16_t* __restrict__ qb,
                  bf16_t* __restrict__ kb, bf16_t* __restrict__ vb) {
    const int bt = blockIdx.x;
    const int b = bt / T_, t = bt % T_;
    const float* row = qkv + (size_t)bt * NQKV_;
    for (int r = threadIdx.x; r < 1792; r += 256) {
        if (r < 1024) {                       // q pairs: 32 heads * 32 pairs
            int h = r >> 5, i = r & 31;
            float re = row[h * 64 + 2 * i], im = row[h * 64 + 2 * i + 1];
            float c = fcos[t * 32 + i], s = fsin[t * 32 + i];
            size_t o = (((size_t)(b * NH_ + h)) * T_ + t) * HS_ + 2 * i;
            qb[o]     = (bf16_t)(re * c - im * s);
            qb[o + 1] = (bf16_t)(re * s + im * c);
        } else if (r < 1280) {                // k pairs: 8 heads * 32 pairs
            int rk = r - 1024;
            int h = rk >> 5, i = rk & 31;
            float re = row[2048 + h * 64 + 2 * i], im = row[2048 + h * 64 + 2 * i + 1];
            float c = fcos[t * 32 + i], s = fsin[t * 32 + i];
            size_t o = (((size_t)(b * NKVH_ + h)) * T_ + t) * HS_ + 2 * i;
            kb[o]     = (bf16_t)(re * c - im * s);
            kb[o + 1] = (bf16_t)(re * s + im * c);
        } else {                              // v copy: 512 elements
            int rv = r - 1280;
            int h = rv >> 6, d = rv & 63;
            size_t o = (((size_t)(b * NKVH_ + h)) * T_ + t) * HS_ + d;
            vb[o] = (bf16_t)row[2560 + rv];
        }
    }
}

// ---------------------------------------------------------------------------
// Flash attention, causal, GQA.  One wave (32 threads) per 16-query tile.
// grid = (T/16, NH, B).  Key block = 32, online softmax in f32.
// ---------------------------------------------------------------------------
__global__ __launch_bounds__(32)
void attn_fa(const bf16_t* __restrict__ qbuf, const bf16_t* __restrict__ kbuf,
             const bf16_t* __restrict__ vbuf, bf16_t* __restrict__ ybuf) {
    __shared__ __align__(16) unsigned short sP[16 * 32];   // P [m][kpos]
    __shared__ __align__(16) unsigned short sVt[64 * 32];  // V^T [d][kpos]

    const int lane = threadIdx.x;
    const int qtile = blockIdx.x, h = blockIdx.y, b = blockIdx.z;
    const int kvh = h / REP_;
    const int lh = lane & 15, hi = lane >> 4;

    const unsigned short* qh = (const unsigned short*)qbuf + (((size_t)(b * NH_ + h)) * T_) * HS_;
    const unsigned short* kh = (const unsigned short*)kbuf + (((size_t)(b * NKVH_ + kvh)) * T_) * HS_;
    const unsigned short* vh = (const unsigned short*)vbuf + (((size_t)(b * NKVH_ + kvh)) * T_) * HS_;

    // Q fragments for the two K=32 chunks of HS=64
    v16bf aq[2];
    #pragma unroll
    for (int c = 0; c < 2; ++c) {
        union { uint4 q[2]; v16bf v; } u;
        const unsigned short* p = qh + (size_t)(qtile * 16 + lh) * HS_ + c * 32 + hi * 8;
        u.q[0] = *(const uint4*)p;
        u.q[1] = *(const uint4*)(p + 16);
        aq[c] = u.v;
    }

    float mstate[8], lstate[8];
    #pragma unroll
    for (int j = 0; j < 8; ++j) { mstate[j] = -1e30f; lstate[j] = 0.0f; }
    v8f acc[4];
    #pragma unroll
    for (int i = 0; i < 4; ++i)
        #pragma unroll
        for (int j = 0; j < 8; ++j) acc[i][j] = 0.0f;

    const int qmax = qtile * 16 + 15;
    for (int kb = 0; kb * 32 <= qmax; ++kb) {
        const int kbase = kb * 32;
        if (kbase + 32 <= qmax) {   // prefetch next K/V block (global_prefetch_b8)
            __builtin_prefetch(kh + (size_t)(kbase + 32 + lane) * HS_, 0, 0);
            __builtin_prefetch(vh + (size_t)(kbase + 32 + lane) * HS_, 0, 0);
        }
        v8f s0, s1;
        #pragma unroll
        for (int j = 0; j < 8; ++j) { s0[j] = 0.0f; s1[j] = 0.0f; }

        // scores: S = Q(16x64) . K^T  over two K=32 chunks, two 16-key tiles
        #pragma unroll
        for (int c = 0; c < 2; ++c) {
            {
                union { uint4 q[2]; v16bf v; } u;
                const unsigned short* p = kh + (size_t)(kbase + lh) * HS_ + c * 32 + hi * 8;
                u.q[0] = *(const uint4*)p;
                u.q[1] = *(const uint4*)(p + 16);
                s0 = __builtin_amdgcn_wmma_f32_16x16x32_bf16(
                    false, aq[c], false, u.v, (short)0, s0, false, false);
            }
            {
                union { uint4 q[2]; v16bf v; } u;
                const unsigned short* p = kh + (size_t)(kbase + 16 + lh) * HS_ + c * 32 + hi * 8;
                u.q[0] = *(const uint4*)p;
                u.q[1] = *(const uint4*)(p + 16);
                s1 = __builtin_amdgcn_wmma_f32_16x16x32_bf16(
                    false, aq[c], false, u.v, (short)0, s1, false, false);
            }
        }

        // scale, causal mask, online softmax update; P -> LDS (bf16)
        #pragma unroll
        for (int j = 0; j < 8; ++j) {
            const int m = j + hi * 8;
            const int qrow = qtile * 16 + m;
            float v0 = s0[j] * 0.125f;
            float v1 = s1[j] * 0.125f;
            if (kbase + lh > qrow)      v0 = -1e30f;
            if (kbase + 16 + lh > qrow) v1 = -1e30f;
            float rm = fmaxf(v0, v1);
            #pragma unroll
            for (int off = 8; off > 0; off >>= 1)
                rm = fmaxf(rm, __shfl_xor(rm, off, 16));
            const float mn = fmaxf(mstate[j], rm);
            const float p0 = __expf(v0 - mn);
            const float p1 = __expf(v1 - mn);
            float rs = p0 + p1;
            #pragma unroll
            for (int off = 8; off > 0; off >>= 1)
                rs += __shfl_xor(rs, off, 16);
            const float scale = __expf(mstate[j] - mn);
            lstate[j] = lstate[j] * scale + rs;
            mstate[j] = mn;
            #pragma unroll
            for (int nt = 0; nt < 4; ++nt) acc[nt][j] *= scale;
            sP[m * 32 + lh]      = f2bf_bits(p0);
            sP[m * 32 + 16 + lh] = f2bf_bits(p1);
        }

        // stage V block 32x64 -> LDS transposed [d][kpos]
        {
            const unsigned short* vp = vh + (size_t)(kbase + lane) * HS_;
            #pragma unroll
            for (int i = 0; i < 8; ++i) {
                union { uint4 q; unsigned short hh[8]; } u;
                u.q = *(const uint4*)(vp + i * 8);
                #pragma unroll
                for (int e = 0; e < 8; ++e)
                    sVt[(i * 8 + e) * 32 + lane] = u.hh[e];
            }
        }

        // P fragment (A layout) from LDS
        union { uint4 q[2]; v16bf v; } up;
        {
            const unsigned short* p = &sP[lh * 32 + hi * 8];
            up.q[0] = *(const uint4*)p;
            up.q[1] = *(const uint4*)(p + 16);
        }
        // PV: acc += P(16x32) . V(32x64)
        #pragma unroll
        for (int nt = 0; nt < 4; ++nt) {
            union { uint4 q[2]; v16bf v; } uv;
            const unsigned short* p = &sVt[(nt * 16 + lh) * 32 + hi * 8];
            uv.q[0] = *(const uint4*)p;
            uv.q[1] = *(const uint4*)(p + 16);
            acc[nt] = __builtin_amdgcn_wmma_f32_16x16x32_bf16(
                false, up.v, false, uv.v, (short)0, acc[nt], false, false);
        }
    }

    // epilogue: y[b, t, h*HS + d] (bf16, row-major over C) for proj GEMM
    #pragma unroll
    for (int nt = 0; nt < 4; ++nt)
        #pragma unroll
        for (int j = 0; j < 8; ++j) {
            const int m = j + hi * 8;
            const int trow = qtile * 16 + m;
            const int d = nt * 16 + lh;
            ybuf[((size_t)(b * T_ + trow)) * C_ + h * HS_ + d] =
                (bf16_t)(acc[nt][j] / lstate[j]);
        }
}

// ---------------------------------------------------------------------------
// Host-side orchestration
// ---------------------------------------------------------------------------
extern "C" void kernel_launch(void* const* d_in, const int* in_sizes, int n_in,
                              void* d_out, int out_size, void* d_ws, size_t ws_size,
                              hipStream_t stream) {
    const float* x     = (const float*)d_in[0];
    const float* fcos  = (const float*)d_in[1];
    const float* fsin  = (const float*)d_in[2];
    const float* Wqkv  = (const float*)d_in[3];
    const float* bqkv  = (const float*)d_in[4];
    const float* Wproj = (const float*)d_in[5];
    const float* bproj = (const float*)d_in[6];
    float* out = (float*)d_out;

    const size_t n_x     = (size_t)B_ * T_ * C_;           // 8,388,608
    const size_t n_wqkv  = (size_t)C_ * NQKV_;             // 6,291,456
    const size_t n_wproj = (size_t)C_ * C_;                // 4,194,304
    const size_t n_qkv   = (size_t)B_ * T_ * NQKV_;        // 12,582,912
    const size_t n_q     = (size_t)B_ * NH_ * T_ * HS_;    // 8,388,608
    const size_t n_kv    = (size_t)B_ * NKVH_ * T_ * HS_;  // 2,097,152

    char* ws = (char*)d_ws;
    size_t off = 0;
    auto carve = [&](size_t bytes) -> void* {
        void* p = ws + off;
        off += (bytes + 255) & ~(size_t)255;
        return p;
    };
    bf16_t* xb     = (bf16_t*)carve(n_x * 2);
    bf16_t* wqkvb  = (bf16_t*)carve(n_wqkv * 2);
    bf16_t* wprojb = (bf16_t*)carve(n_wproj * 2);
    float*  qkvf   = (float*)carve(n_qkv * 4);
    bf16_t* qb     = (bf16_t*)carve(n_q * 2);
    bf16_t* kb     = (bf16_t*)carve(n_kv * 2);
    bf16_t* vb     = (bf16_t*)carve(n_kv * 2);
    bf16_t* yb     = (bf16_t*)carve(n_x * 2);
    (void)ws_size; (void)in_sizes; (void)n_in; (void)out_size;

    // 1) convert inputs to bf16
    f32_to_bf16_kernel<<<(unsigned)((n_x + 255) / 256), 256, 0, stream>>>(x, xb, n_x);
    f32_to_bf16_kernel<<<(unsigned)((n_wqkv + 255) / 256), 256, 0, stream>>>(Wqkv, wqkvb, n_wqkv);
    f32_to_bf16_kernel<<<(unsigned)((n_wproj + 255) / 256), 256, 0, stream>>>(Wproj, wprojb, n_wproj);

    // 2) qkv = x @ Wqkv + bqkv   (M=4096, N=3072, K=2048)
    gemm_bf16_wmma<<<dim3(B_ * T_ / 128, NQKV_ / 128), 256, 0, stream>>>(
        xb, wqkvb, bqkv, qkvf, B_ * T_, NQKV_, C_);

    // 3) RoPE + scatter to head-major bf16 q/k/v
    rope_scatter<<<B_ * T_, 256, 0, stream>>>(qkvf, fcos, fsin, qb, kb, vb);

    // 4) causal GQA flash attention -> y bf16 [B*T, C]
    attn_fa<<<dim3(T_ / 16, NH_, B_), 32, 0, stream>>>(qb, kb, vb, yb);

    // 5) out = y @ Wproj + bproj   (M=4096, N=2048, K=2048)
    gemm_bf16_wmma<<<dim3(B_ * T_ / 128, C_ / 128), 256, 0, stream>>>(
        yb, wprojb, bproj, out, B_ * T_, C_, C_);
}
